// YOLOF_model_32100585571110
// MI455X (gfx1250) — compile-verified, matrix operand order checked by reading, not run on hardware
//
#include <hip/hip_runtime.h>
#include <hip/hip_bf16.h>
#include <math.h>

typedef __attribute__((ext_vector_type(16))) __bf16 v16bf;
typedef __attribute__((ext_vector_type(8)))  float  v8f;

#define NB   16
#define HH   32
#define WW   32
#define PIX  (NB*HH*WW)     // 16384 pixels
#define AA   9
#define CC   80
#define ACC_CH (AA*CC)      // 720
#define SCALE_CLAMP 4.135166556742356f
#define INFV 100000000.0f

// LDS layout (dynamic shared, double buffered):
//   buffer b at b*20480:  A tile 128 rows x 80B (32 bf16 + pad) = 10240B
//                         B tile 128 rows x 80B                 = 10240B
#define LDS_BUF_STRIDE 20480u
#define LDS_B_OFF      10240u
#define LDS_TOTAL      (2 * LDS_BUF_STRIDE)

// ---------------------------------------------------------------------------
// CDNA5 async global->LDS copy (ASYNCcnt-tracked) + wait
// ---------------------------------------------------------------------------
__device__ __forceinline__ void async_ld_b128(unsigned ldsoff, unsigned long long gaddr) {
    asm volatile("global_load_async_to_lds_b128 %0, %1, off"
                 :: "v"(ldsoff), "v"(gaddr) : "memory");
}
__device__ __forceinline__ void wait_async_0() {
    asm volatile("s_wait_asynccnt 0x0" ::: "memory");
}

// ---------------------------------------------------------------------------
// f32 -> bf16 convert
// ---------------------------------------------------------------------------
__global__ void cvt_bf16_k(const float* __restrict__ src, __bf16* __restrict__ dst, long n) {
    long i = (long)blockIdx.x * 256 + threadIdx.x;
    if (i < n) dst[i] = (__bf16)src[i];
}

// ---------------------------------------------------------------------------
// weight pack: src f32 [tap][Cin][Cout] -> dst bf16 [Cout][tap][Cin]
// ---------------------------------------------------------------------------
__global__ void pack_w_k(const float* __restrict__ src, __bf16* __restrict__ dst,
                         int taps, int Cin, int Cout) {
    long n = (long)taps * Cin * Cout;
    long i = (long)blockIdx.x * 256 + threadIdx.x;
    if (i >= n) return;
    long tc = (long)taps * Cin;
    int co = (int)(i / tc);
    int r  = (int)(i - (long)co * tc);          // tap*Cin + ci
    dst[i] = (__bf16)src[(long)r * Cout + co];
}

// ---------------------------------------------------------------------------
// implicit-GEMM conv, WMMA bf16 -> f32, double-buffered async LDS staging.
//   block tile: 128 pixels (M) x 128 out-channels (N), 8 waves (4 M x 2 N),
//   each wave: 2 A frags (32x32) x 4 B frags (32x64) -> 8 v_wmma per K step.
//   epilogue: y = conv*scale + bias ; optional relu ; optional (+residual).
// ---------------------------------------------------------------------------
__global__ __launch_bounds__(256)
void conv_wmma_k(const __bf16* __restrict__ in, const __bf16* __restrict__ wpack,
                 const float* __restrict__ scale, const float* __restrict__ bias,
                 const __bf16* __restrict__ resid,
                 __bf16* __restrict__ out_bf, float* __restrict__ out_f32,
                 int Cin, int Cout, int ksize, int dil, int relu)
{
    extern __shared__ __align__(16) char smem[];   // LDS_TOTAL bytes

    const int t     = threadIdx.x;
    const int lane  = t & 31;
    const int wv    = t >> 5;
    const int mw    = wv & 3;         // 0..3  (32-pixel strip)
    const int nw    = wv >> 2;        // 0..1  (64-channel strip)
    const int lrow  = lane & 15;
    const int khalf = lane >> 4;

    const int m0 = blockIdx.x * 128;
    const int n0 = blockIdx.y * 128;

    const int taps = ksize * ksize;
    const int kcn  = Cin >> 5;                 // 32-wide K chunks (power of 2)
    const int ksh  = __popc(kcn - 1);          // log2(kcn)
    const int KS   = taps * kcn;

    v8f acc[2][4] = {};

    // ---- stage one 128x32 A tile + 128x32 B tile into LDS buffer `buf`
    auto stage = [&](int ks, int buf) {
        const int tap = ks >> ksh;
        const int kc  = ks & (kcn - 1);
        const int ky  = (ksize == 3) ? (tap / 3 - 1) : 0;
        const int kx  = (ksize == 3) ? (tap % 3 - 1) : 0;
        const unsigned base = (unsigned)buf * LDS_BUF_STRIDE;
        // A: gathered activations (async; zero-fill OOB with plain LDS stores)
        #pragma unroll
        for (int j = 0; j < 2; ++j) {
            const int c   = t + j * 256;
            const int pix = c >> 2;
            const int seg = c & 3;
            const int gp  = m0 + pix;
            const int img = gp >> 10;
            const int rem = gp & 1023;
            const int sy  = (rem >> 5) + ky * dil;
            const int sx  = (rem & 31) + kx * dil;
            const unsigned lo = base + (unsigned)pix * 80u + (unsigned)seg * 16u;
            if ((sy >= 0) & (sy < HH) & (sx >= 0) & (sx < WW)) {
                const __bf16* g = in + (long)((img << 10) + (sy << 5) + sx) * Cin
                                     + (kc << 5) + seg * 8;
                async_ld_b128(lo, (unsigned long long)(uintptr_t)g);
            } else {
                *(uint4*)(smem + lo) = make_uint4(0u, 0u, 0u, 0u);
            }
        }
        // B: packed weights [Cout][tap][Cin]
        #pragma unroll
        for (int j = 0; j < 2; ++j) {
            const int c   = t + j * 256;
            const int col = c >> 2;
            const int seg = c & 3;
            const int co  = n0 + col;
            const unsigned lo = base + LDS_B_OFF + (unsigned)col * 80u + (unsigned)seg * 16u;
            if (co < Cout) {
                const __bf16* g = wpack + ((long)co * taps + tap) * Cin
                                        + (kc << 5) + seg * 8;
                async_ld_b128(lo, (unsigned long long)(uintptr_t)g);
            } else {
                *(uint4*)(smem + lo) = make_uint4(0u, 0u, 0u, 0u);
            }
        }
    };

    stage(0, 0);
    wait_async_0();
    __syncthreads();

    int cur = 0;
    for (int ks = 0; ks < KS; ++ks) {
        if (ks + 1 < KS) stage(ks + 1, cur ^ 1);   // async prefetch next tile

        // ---- compute on current tile
        const unsigned ab = (unsigned)cur * LDS_BUF_STRIDE;
        union FU { v16bf v; uint4 u[2]; } a[2];
        #pragma unroll
        for (int am = 0; am < 2; ++am) {
            const char* p = smem + ab + (unsigned)(mw * 32 + am * 16 + lrow) * 80u;
            a[am].u[0] = *(const uint4*)(p + khalf * 16);
            a[am].u[1] = *(const uint4*)(p + 32 + khalf * 16);
        }
        #pragma unroll
        for (int bb = 0; bb < 4; ++bb) {
            union FU b;
            const char* p = smem + ab + LDS_B_OFF + (unsigned)(nw * 64 + bb * 16 + lrow) * 80u;
            b.u[0] = *(const uint4*)(p + khalf * 16);
            b.u[1] = *(const uint4*)(p + 32 + khalf * 16);
            #pragma unroll
            for (int am = 0; am < 2; ++am)
                acc[am][bb] = __builtin_amdgcn_wmma_f32_16x16x32_bf16(
                    false, a[am].v, false, b.v, (short)0, acc[am][bb], false, false);
        }

        wait_async_0();       // next tile's async DMA complete
        __syncthreads();      // all waves done reading cur / writing alt
        cur ^= 1;
    }

    // ---- epilogue: C/D layout: VGPR r, lanes0-15 -> M=r, lanes16-31 -> M=r+8
    #pragma unroll
    for (int bb = 0; bb < 4; ++bb) {
        const int co = n0 + nw * 64 + bb * 16 + lrow;
        if (co >= Cout) continue;
        const float s = scale ? scale[co] : 1.0f;
        const float b = bias  ? bias[co]  : 0.0f;
        #pragma unroll
        for (int am = 0; am < 2; ++am) {
            const int pb = m0 + mw * 32 + am * 16 + 8 * khalf;
            #pragma unroll
            for (int r = 0; r < 8; ++r) {
                const int p = pb + r;
                float v = acc[am][bb][r] * s + b;
                if (relu) v = v > 0.0f ? v : 0.0f;
                if (resid) v += (float)resid[(long)p * Cout + co];
                const long o = (long)p * Cout + co;
                if (out_bf) out_bf[o] = (__bf16)v;
                else        out_f32[o] = v;
            }
        }
    }
}

// ---------------------------------------------------------------------------
// norm_cls = cls + obj - log(1 + min(exp(cls),INF) + min(exp(obj),INF))
// in-place on d_out (cls_score written there by score conv)
// ---------------------------------------------------------------------------
__global__ void norm_cls_k(float* __restrict__ cls_io, const float* __restrict__ obj, long n) {
    long i = (long)blockIdx.x * 256 + threadIdx.x;
    if (i >= n) return;
    long p  = i / ACC_CH;
    int rem = (int)(i - p * ACC_CH);
    int a   = rem / CC;
    float cs = cls_io[i];
    float ob = obj[p * AA + a];
    float e1 = fminf(__expf(cs), INFV);
    float e2 = fminf(__expf(ob), INFV);
    cls_io[i] = cs + ob - __logf(1.0f + e1 + e2);
}

// ---------------------------------------------------------------------------
// anchor decode
// ---------------------------------------------------------------------------
__global__ void boxes_k(const float* __restrict__ breg, float* __restrict__ outp) {
    int idx = blockIdx.x * 256 + threadIdx.x;       // PIX * AA = 147456
    if (idx >= PIX * AA) return;
    int p = idx / AA, a = idx - p * AA;
    int rem = p & 1023;
    int py = rem >> 5, px = rem & 31;
    float cx = (px + 0.5f) * 8.0f;
    float cy = (py + 0.5f) * 8.0f;
    int ridx = a / 3, sidx = a - ridx * 3;
    float ratio = (ridx == 0) ? 0.5f : ((ridx == 1) ? 1.0f : 2.0f);
    float ah = sqrtf(1024.0f / ratio);
    float aw = 1024.0f / ah;
    float sc = exp2f((float)sidx * (1.0f / 3.0f));
    float wa = sc * aw, ha = sc * ah;
    const float* g = breg + (long)p * (AA * 4) + a * 4;
    float dx = g[0], dy = g[1];
    float dw = fminf(g[2], SCALE_CLAMP);
    float dh = fminf(g[3], SCALE_CLAMP);
    float pcx = dx * wa + cx;
    float pcy = dy * ha + cy;
    float pw = __expf(dw) * wa;
    float ph = __expf(dh) * ha;
    float* o = outp + (long)idx * 4;
    o[0] = pcx - 0.5f * pw;
    o[1] = pcy - 0.5f * ph;
    o[2] = pcx + 0.5f * pw;
    o[3] = pcy + 0.5f * ph;
}

// ---------------------------------------------------------------------------
extern "C" void kernel_launch(void* const* d_in, const int* in_sizes, int n_in,
                              void* d_out, int out_size, void* d_ws, size_t ws_size,
                              hipStream_t stream)
{
    const float* x       = (const float*)d_in[0];
    const float* enc_w1  = (const float*)d_in[1];
    const float* enc_s1  = (const float*)d_in[2];
    const float* enc_b1  = (const float*)d_in[3];
    const float* enc_w2  = (const float*)d_in[4];
    const float* enc_s2  = (const float*)d_in[5];
    const float* enc_b2  = (const float*)d_in[6];
    const float* blk_w1  = (const float*)d_in[7];
    const float* blk_s1  = (const float*)d_in[8];
    const float* blk_b1  = (const float*)d_in[9];
    const float* blk_w2  = (const float*)d_in[10];
    const float* blk_s2  = (const float*)d_in[11];
    const float* blk_b2  = (const float*)d_in[12];
    const float* blk_w3  = (const float*)d_in[13];
    const float* blk_s3  = (const float*)d_in[14];
    const float* blk_b3  = (const float*)d_in[15];
    const float* cls_w   = (const float*)d_in[16];
    const float* cls_s   = (const float*)d_in[17];
    const float* cls_b   = (const float*)d_in[18];
    const float* reg_w   = (const float*)d_in[19];
    const float* reg_s   = (const float*)d_in[20];
    const float* reg_b   = (const float*)d_in[21];
    const float* score_w = (const float*)d_in[22];
    const float* score_b = (const float*)d_in[23];
    const float* bbox_w  = (const float*)d_in[24];
    const float* bbox_b  = (const float*)d_in[25];
    const float* obj_w   = (const float*)d_in[26];
    const float* obj_b   = (const float*)d_in[27];
    (void)in_sizes; (void)n_in; (void)out_size; (void)ws_size;

    char* ws = (char*)d_ws;
    size_t off = 0;
    auto alloc = [&](size_t bytes) -> void* {
        void* p = ws + off;
        off = (off + bytes + 255) & ~(size_t)255;
        return p;
    };

    __bf16* x_bf  = (__bf16*)alloc((size_t)PIX * 2048 * 2);
    __bf16* buf0  = (__bf16*)alloc((size_t)PIX * 512 * 2);
    __bf16* buf1  = (__bf16*)alloc((size_t)PIX * 512 * 2);
    __bf16* buf2  = (__bf16*)alloc((size_t)PIX * 512 * 2);
    __bf16* t128a = (__bf16*)alloc((size_t)PIX * 128 * 2);
    __bf16* t128b = (__bf16*)alloc((size_t)PIX * 128 * 2);
    float*  bbox_f = (float*)alloc((size_t)PIX * AA * 4 * 4);
    float*  obj_f  = (float*)alloc((size_t)PIX * AA * 4);

    __bf16* w_enc1  = (__bf16*)alloc((size_t)2048 * 512 * 2);
    __bf16* w_enc2  = (__bf16*)alloc((size_t)9 * 512 * 512 * 2);
    __bf16* w_b1[4], *w_b2[4], *w_b3[4];
    for (int i = 0; i < 4; ++i) {
        w_b1[i] = (__bf16*)alloc((size_t)512 * 128 * 2);
        w_b2[i] = (__bf16*)alloc((size_t)9 * 128 * 128 * 2);
        w_b3[i] = (__bf16*)alloc((size_t)128 * 512 * 2);
    }
    __bf16* w_cls[2];
    for (int i = 0; i < 2; ++i) w_cls[i] = (__bf16*)alloc((size_t)9 * 512 * 512 * 2);
    __bf16* w_reg[4];
    for (int i = 0; i < 4; ++i) w_reg[i] = (__bf16*)alloc((size_t)9 * 512 * 512 * 2);
    __bf16* w_score = (__bf16*)alloc((size_t)9 * 512 * ACC_CH * 2);
    __bf16* w_bbox  = (__bf16*)alloc((size_t)9 * 512 * (AA * 4) * 2);
    __bf16* w_obj   = (__bf16*)alloc((size_t)9 * 512 * AA * 2);

    // --- prep: converts + weight packs
    {
        long n = (long)PIX * 2048;
        cvt_bf16_k<<<dim3((unsigned)((n + 255) / 256)), dim3(256), 0, stream>>>(x, x_bf, n);
    }
    auto pack = [&](const float* src, __bf16* dst, int taps, int Cin, int Cout) {
        long n = (long)taps * Cin * Cout;
        pack_w_k<<<dim3((unsigned)((n + 255) / 256)), dim3(256), 0, stream>>>(src, dst, taps, Cin, Cout);
    };
    pack(enc_w1, w_enc1, 1, 2048, 512);
    pack(enc_w2, w_enc2, 9, 512, 512);
    for (int i = 0; i < 4; ++i) {
        pack(blk_w1 + (long)i * 512 * 128,     w_b1[i], 1, 512, 128);
        pack(blk_w2 + (long)i * 9 * 128 * 128, w_b2[i], 9, 128, 128);
        pack(blk_w3 + (long)i * 128 * 512,     w_b3[i], 1, 128, 512);
    }
    for (int i = 0; i < 2; ++i) pack(cls_w + (long)i * 9 * 512 * 512, w_cls[i], 9, 512, 512);
    for (int i = 0; i < 4; ++i) pack(reg_w + (long)i * 9 * 512 * 512, w_reg[i], 9, 512, 512);
    pack(score_w, w_score, 9, 512, ACC_CH);
    pack(bbox_w,  w_bbox,  9, 512, AA * 4);
    pack(obj_w,   w_obj,   9, 512, AA);

    auto conv = [&](const __bf16* in, const __bf16* wp, const float* s, const float* b,
                    const __bf16* resid, __bf16* outb, float* outf,
                    int Cin, int Cout, int ks, int dil, int relu) {
        dim3 g(PIX / 128, (unsigned)((Cout + 127) / 128));
        conv_wmma_k<<<g, dim3(256), LDS_TOTAL, stream>>>(in, wp, s, b, resid, outb, outf,
                                                         Cin, Cout, ks, dil, relu);
    };

    // --- encoder
    conv(x_bf, w_enc1, enc_s1, enc_b1, nullptr, buf0, nullptr, 2048, 512, 1, 1, 0);
    conv(buf0, w_enc2, enc_s2, enc_b2, nullptr, buf1, nullptr, 512, 512, 3, 1, 0);

    // --- dilated residual blocks
    __bf16* ycur = buf1;
    __bf16* yalt = buf2;
    const int dils[4] = {2, 4, 6, 8};
    for (int i = 0; i < 4; ++i) {
        conv(ycur,  w_b1[i], blk_s1 + i * 128, blk_b1 + i * 128, nullptr, t128a, nullptr, 512, 128, 1, 1, 1);
        conv(t128a, w_b2[i], blk_s2 + i * 128, blk_b2 + i * 128, nullptr, t128b, nullptr, 128, 128, 3, dils[i], 1);
        conv(t128b, w_b3[i], blk_s3 + i * 512, blk_b3 + i * 512, ycur,    yalt,  nullptr, 128, 512, 1, 1, 1);
        __bf16* tmp = ycur; ycur = yalt; yalt = tmp;
    }
    // after 4 swaps: ycur == buf1, yalt == buf2

    // --- cls tower + score head (score conv writes f32 straight into d_out)
    conv(ycur, w_cls[0], cls_s,       cls_b,       nullptr, buf0, nullptr, 512, 512, 3, 1, 1);
    conv(buf0, w_cls[1], cls_s + 512, cls_b + 512, nullptr, buf2, nullptr, 512, 512, 3, 1, 1);
    conv(buf2, w_score, nullptr, score_b, nullptr, nullptr, (float*)d_out, 512, ACC_CH, 3, 1, 0);

    // --- reg tower + bbox/obj heads
    conv(ycur, w_reg[0], reg_s,          reg_b,          nullptr, buf0, nullptr, 512, 512, 3, 1, 1);
    conv(buf0, w_reg[1], reg_s + 512,    reg_b + 512,    nullptr, buf2, nullptr, 512, 512, 3, 1, 1);
    conv(buf2, w_reg[2], reg_s + 2*512,  reg_b + 2*512,  nullptr, buf0, nullptr, 512, 512, 3, 1, 1);
    conv(buf0, w_reg[3], reg_s + 3*512,  reg_b + 3*512,  nullptr, buf2, nullptr, 512, 512, 3, 1, 1);
    conv(buf2, w_bbox, nullptr, bbox_b, nullptr, nullptr, bbox_f, 512, AA * 4, 3, 1, 0);
    conv(buf2, w_obj,  nullptr, obj_b,  nullptr, nullptr, obj_f,  512, AA,     3, 1, 0);

    // --- post-processing
    {
        long n = (long)PIX * ACC_CH;
        norm_cls_k<<<dim3((unsigned)((n + 255) / 256)), dim3(256), 0, stream>>>((float*)d_out, obj_f, n);
    }
    {
        int n = PIX * AA;
        float* box_out = (float*)d_out + (long)PIX * ACC_CH;
        boxes_k<<<dim3((unsigned)((n + 255) / 256)), dim3(256), 0, stream>>>(bbox_f, box_out);
    }
}
